// VisionTransformerBlock_34144990003454
// MI455X (gfx1250) — matrix operand order repaired and checked by reading, not compiled, not run on hardware
//
#include <hip/hip_runtime.h>
#include <hip/hip_bf16.h>
#include <math.h>

typedef __attribute__((ext_vector_type(16))) __bf16 v16bf;
typedef __attribute__((ext_vector_type(8)))  float  v8f;
typedef __attribute__((ext_vector_type(4)))  unsigned int v4u;
typedef __attribute__((ext_vector_type(8)))  int  v8i;
typedef __attribute__((ext_vector_type(4)))  int  v4i;

#if __has_builtin(__builtin_amdgcn_tensor_load_to_lds)
#define USE_TDM 1
#else
#define USE_TDM 0
#endif

// ---------------------------------------------------------------------------
// helpers
// ---------------------------------------------------------------------------
__device__ __forceinline__ __bf16 f2bf(float f) {
    unsigned u = __float_as_uint(f);
    unsigned r = (u + 0x7FFFu + ((u >> 16) & 1u)) >> 16;
    unsigned short s = (unsigned short)r;
    __bf16 h;
    __builtin_memcpy(&h, &s, 2);
    return h;
}

__device__ __forceinline__ float gelu_exact(float v) {
    return 0.5f * v * (1.0f + erff(v * 0.70710678118654752f));
}

// Build a v16bf WMMA fragment from two 16-byte chunks (lo = elements 0..7,
// hi = elements 8..15). Matches the CDNA5 16-bit A/B fragment layouts when
// the source rows are K-contiguous.
__device__ __forceinline__ v16bf frag16(const __bf16* lo, const __bf16* hi) {
    uint4 t[2];
    t[0] = *(const uint4*)lo;
    t[1] = *(const uint4*)hi;
    v16bf r;
    __builtin_memcpy(&r, t, 32);
    return r;
}

#if USE_TDM
// Issue a TDM load of a 2-D bf16 tile (tile_rows x tile_cols, row-major,
// row stride = stride_elems) from global memory into LDS. Rows >= rows_avail
// are zero-filled by TDM OOB handling. Wave-level op (EXEC ignored).
__device__ __forceinline__ void tdm_load_tile_bf16(
    unsigned lds_addr, const __bf16* gptr, int rows_avail,
    unsigned tile_rows, unsigned tile_cols, unsigned stride_elems)
{
    unsigned long long ga = (unsigned long long)(const void*)gptr;
    unsigned td0 = tile_cols;                       // exact bound along dim0
    unsigned td1 = (rows_avail > 0) ? (unsigned)rows_avail : 0u;
    v4u g0;
    g0[0] = 1u;                                     // count=1, user-mode load
    g0[1] = lds_addr;                               // LDS byte address
    g0[2] = (unsigned)(ga & 0xFFFFFFFFull);         // global_addr[31:0]
    g0[3] = (unsigned)((ga >> 32) & 0x01FFFFFFull)  // global_addr[56:32]
            | (2u << 30);                           // type = 2 ("image")
    v8i g1;
    g1[0] = (int)(1u << 16);                        // data_size = 1 -> 2 bytes
    g1[1] = (int)((td0 & 0xFFFFu) << 16);           // tensor_dim0[15:0]
    g1[2] = (int)((td0 >> 16) | ((td1 & 0xFFFFu) << 16));
    g1[3] = (int)((td1 >> 16) | (tile_cols << 16)); // tile_dim0
    g1[4] = (int)tile_rows;                         // tile_dim1 (tile_dim2=0)
    g1[5] = (int)stride_elems;                      // tensor_dim0_stride[31:0]
    g1[6] = 0;                                      // stride0 hi / stride1 lo
    g1[7] = 0;
    v4i g2 = {0, 0, 0, 0};
    v4i g3 = {0, 0, 0, 0};
#if __has_include(<hip/amd_detail/amd_gfx1250_TDM.h>)
    v8i g4 = {0, 0, 0, 0, 0, 0, 0, 0};
    __builtin_amdgcn_tensor_load_to_lds(g0, g1, g2, g3, g4, 0);
#else
    __builtin_amdgcn_tensor_load_to_lds(g0, g1, g2, g3, 0);
#endif
}
#endif

// ---------------------------------------------------------------------------
// fp32 -> bf16 flat convert (for weights)
// ---------------------------------------------------------------------------
__global__ void k_cvt(const float* __restrict__ in, __bf16* __restrict__ out, int n) {
    int i = blockIdx.x * blockDim.x + threadIdx.x;
    if (i < n) out[i] = f2bf(in[i]);
}

// ---------------------------------------------------------------------------
// Generic LayerNorm over last dim D -> bf16 output. One block per row.
// ---------------------------------------------------------------------------
__global__ void k_ln(const float* __restrict__ in, const float* __restrict__ g,
                     const float* __restrict__ bt, __bf16* __restrict__ out, int D) {
    int row = blockIdx.x;
    const float* p = in + (size_t)row * D;
    __shared__ float s1[256];
    __shared__ float s2[256];
    int t = threadIdx.x, nt = blockDim.x;
    float sum = 0.f, sq = 0.f;
    for (int i = t; i < D; i += nt) { float v = p[i]; sum += v; sq += v * v; }
    s1[t] = sum; s2[t] = sq;
    __syncthreads();
    for (int o = nt >> 1; o > 0; o >>= 1) {
        if (t < o) { s1[t] += s1[t + o]; s2[t] += s2[t + o]; }
        __syncthreads();
    }
    float mean = s1[0] / (float)D;
    float var  = s2[0] / (float)D - mean * mean;
    float inv  = rsqrtf(var + 1e-5f);
    for (int i = t; i < D; i += nt) {
        float v = (p[i] - mean) * inv * g[i] + bt[i];
        out[(size_t)row * D + i] = f2bf(v);
    }
}

// ---------------------------------------------------------------------------
// Tiled WMMA GEMM:  out = epilogue(A[M,K] @ B[K,N] + bias [+ res])
// A,B bf16 row-major. BM=64 BN=128 BK=32. 8 waves: 2(M) x 4(N), 32x32 each.
// A tile staged by the Tensor Data Mover (double-buffered, TENSORcnt);
// B tile staged transposed in LDS so all fragment loads are ds_load_b128.
// ---------------------------------------------------------------------------
#define BM 64
#define BN 128
#define BK 32
#define BKP 40   // padded LDS row (keeps 16B alignment, staggers banks)

__global__ __launch_bounds__(256) void k_gemm(
    const __bf16* __restrict__ A, const __bf16* __restrict__ Bw,
    const float* __restrict__ bias, const float* __restrict__ res,
    float* __restrict__ outF, __bf16* __restrict__ outB,
    int M, int N, int K, int act)
{
#if USE_TDM
    __shared__ __align__(16) __bf16 sA[2][BM][BK];  // 8 KB, double-buffered
#else
    __shared__ __align__(16) __bf16 sA[1][BM][BK];  // 4 KB
#endif
    __shared__ __align__(16) __bf16 sBt[BN][BKP];   // 10 KB, transposed

    const int tid  = threadIdx.x;
    const int wave = tid >> 5;
    const int lane = tid & 31;
    const int half = lane >> 4;
    const int l16  = lane & 15;
    const int wm   = wave >> 2;      // 0..1
    const int wn   = wave & 3;       // 0..3
    const int row0 = blockIdx.x * BM;
    const int col0 = blockIdx.y * BN;

    // staging assignments (each thread: 2 x b128 of B; A handled by TDM)
    const int aR = (tid * 8) >> 5, aC = (tid * 8) & 31;
    const int bR0 = (tid * 8) >> 7,          bC0 = (tid * 8) & 127;
    const int bR1 = ((tid + 256) * 8) >> 7,  bC1 = ((tid + 256) * 8) & 127;

    uint4 aReg, bReg0, bReg1;
    bReg0 = *(const uint4*)(Bw + (size_t)bR0 * N + col0 + bC0);
    bReg1 = *(const uint4*)(Bw + (size_t)bR1 * N + col0 + bC1);
#if USE_TDM
    if (wave == 0) {
        tdm_load_tile_bf16((unsigned)(unsigned long long)(void*)&sA[0][0][0],
                           A + (size_t)row0 * K, M - row0, BM, BK, (unsigned)K);
    }
#else
    {
        int gr = row0 + aR;
        if (gr < M) aReg = *(const uint4*)(A + (size_t)gr * K + aC);
        else { uint4 z = {0, 0, 0, 0}; aReg = z; }
    }
#endif

    v8f acc[2][2] = {};

    for (int k0 = 0; k0 < K; k0 += BK) {
        const int cur = (k0 / BK) & (USE_TDM ? 1 : 0);
        // commit staged B registers to LDS (transposed via 2B scatter)
        {
            __align__(16) __bf16 tmp[8];
            *(uint4*)tmp = bReg0;
            #pragma unroll
            for (int e = 0; e < 8; ++e) sBt[bC0 + e][bR0] = tmp[e];
            *(uint4*)tmp = bReg1;
            #pragma unroll
            for (int e = 0; e < 8; ++e) sBt[bC1 + e][bR1] = tmp[e];
        }
#if USE_TDM
        if (wave == 0) __builtin_amdgcn_s_wait_tensorcnt(0);  // sA[cur] ready
#else
        *(uint4*)&sA[0][aR][aC] = aReg;
#endif
        __syncthreads();

        // issue next tile's loads (overlap with WMMA below)
        const int kn = k0 + BK;
        if (kn < K) {
            bReg0 = *(const uint4*)(Bw + (size_t)(kn + bR0) * N + col0 + bC0);
            bReg1 = *(const uint4*)(Bw + (size_t)(kn + bR1) * N + col0 + bC1);
#if USE_TDM
            if (wave == 0) {
                tdm_load_tile_bf16(
                    (unsigned)(unsigned long long)(void*)&sA[cur ^ 1][0][0],
                    A + (size_t)row0 * K + kn, M - row0, BM, BK, (unsigned)K);
            }
#else
            int gr = row0 + aR;
            if (gr < M) aReg = *(const uint4*)(A + (size_t)gr * K + kn + aC);
            else { uint4 z = {0, 0, 0, 0}; aReg = z; }
#endif
        }

        // fragments: two b128 loads each
        v16bf af[2], bfr[2];
        #pragma unroll
        for (int mf = 0; mf < 2; ++mf) {
            const __bf16* ar = &sA[cur][wm * 32 + mf * 16 + l16][0];
            af[mf] = frag16(ar + half * 8, ar + 16 + half * 8);
        }
        #pragma unroll
        for (int nf = 0; nf < 2; ++nf) {
            const __bf16* br = &sBt[wn * 32 + nf * 16 + l16][0];
            bfr[nf] = frag16(br + half * 8, br + 16 + half * 8);
        }
        #pragma unroll
        for (int mf = 0; mf < 2; ++mf)
            #pragma unroll
            for (int nf = 0; nf < 2; ++nf)
                acc[mf][nf] = __builtin_amdgcn_wmma_f32_16x16x32_bf16(
                    false, af[mf], false, bfr[nf], (short)0, acc[mf][nf], false, false);
        __syncthreads();
    }

    // epilogue
    #pragma unroll
    for (int mf = 0; mf < 2; ++mf)
        #pragma unroll
        for (int nf = 0; nf < 2; ++nf)
            #pragma unroll
            for (int r = 0; r < 8; ++r) {
                int grow = row0 + wm * 32 + mf * 16 + r + 8 * half;
                int gcol = col0 + wn * 32 + nf * 16 + l16;
                if (grow < M) {
                    float v = acc[mf][nf][r] + bias[gcol];
                    if (res)  v += res[(size_t)grow * N + gcol];
                    if (outF) outF[(size_t)grow * N + gcol] = v;
                    if (outB) {
                        float w = act ? gelu_exact(v) : v;
                        outB[(size_t)grow * N + gcol] = f2bf(w);
                    }
                }
            }
}

// ---------------------------------------------------------------------------
// Flash attention with 2-D ALiBi. Grid: (B*H, ceil(N/128)). 8 waves / block,
// each wave owns a 16-row Q tile. Key blocks of 32 staged in LDS; K tile
// copied with CDNA5 async global->LDS loads (ASYNCcnt).
// ---------------------------------------------------------------------------
#define HEADS 12
#define HDIM  64

__global__ __launch_bounds__(256) void k_attn(
    const __bf16* __restrict__ qb, const __bf16* __restrict__ kb,
    const __bf16* __restrict__ vb, const float* __restrict__ positions,
    const float* __restrict__ patch_sizes, const int* __restrict__ nregp,
    __bf16* __restrict__ ob, int Ntok)
{
    const int bh = blockIdx.x;
    const int b  = bh / HEADS;
    const int h  = bh % HEADS;
    const int tid  = threadIdx.x;
    const int wave = tid >> 5;
    const int lane = tid & 31;
    const int half = lane >> 4;
    const int l16  = lane & 15;
    const int qbase = blockIdx.y * 128 + wave * 16;

    __shared__ __align__(16) __bf16 sK[32][HDIM];   // 4 KB, hd-contiguous
    __shared__ __align__(16) __bf16 sVt[HDIM][40];  // 5 KB, transposed
    __shared__ float sPos[32][2];
    __shared__ __align__(16) __bf16 sP[8][16][32];  // per-wave P staging, 8 KB

    const int nreg = *nregp;
    const float slope = (h < 8) ? exp2f(-(float)(h + 1))
                                : exp2f(-0.5f * (float)(2 * (h - 8) + 1));
    const float ps = fmaxf(patch_sizes[b], 1e-6f);
    const float scale = 0.125f; // 1/sqrt(64)

    // Q fragments (16x64 -> two 16x32 A frags) as 4 x global b128 per lane
    v16bf qa0, qa1;
    {
        int qrow = qbase + l16;
        if (qrow >= Ntok) qrow = Ntok - 1;        // clamp; rows never stored
        const __bf16* qp = qb + ((size_t)(b * Ntok + qrow)) * 768 + h * HDIM;
        qa0 = frag16(qp + half * 8,      qp + 16 + half * 8);
        qa1 = frag16(qp + 32 + half * 8, qp + 48 + half * 8);
    }

    // per-lane Q-row metadata (rows r + 8*half for r=0..7)
    float qpx[8], qpy[8];
    int keepq[8];
    #pragma unroll
    for (int r = 0; r < 8; ++r) {
        int row = qbase + r + 8 * half;
        bool v = row < Ntok;
        qpx[r] = v ? positions[((size_t)(b * Ntok + row)) * 2 + 0] : 0.f;
        qpy[r] = v ? positions[((size_t)(b * Ntok + row)) * 2 + 1] : 0.f;
        keepq[r] = (v && !(row == 0 || (row >= 1 && row < 1 + nreg))) ? 1 : 0;
    }

    v8f accO[4] = {};
    float m[8], l[8];
    #pragma unroll
    for (int r = 0; r < 8; ++r) { m[r] = -INFINITY; l[r] = 0.f; }

    const int nkb = (Ntok + 31) / 32;
    for (int kbi = 0; kbi < nkb; ++kbi) {
        const int kb0 = kbi * 32;
        __syncthreads();
        // cooperative stage: K via async global->LDS b128, V transposed
        {
            int idx = tid * 8;
            int r = idx >> 6, c = idx & 63;
            int key = kb0 + r;
            if (key < Ntok) {
                size_t base = ((size_t)(b * Ntok + key)) * 768 + h * HDIM + c;
                unsigned ldsa = (unsigned)(unsigned long long)(const void*)&sK[r][c];
                unsigned long long ga = (unsigned long long)(const void*)(kb + base);
                asm volatile("global_load_async_to_lds_b128 %0, %1, off"
                             :: "v"(ldsa), "v"(ga) : "memory");
                __align__(16) __bf16 tmp[8];
                *(uint4*)tmp = *(const uint4*)(vb + base);
                #pragma unroll
                for (int e = 0; e < 8; ++e) sVt[c + e][r] = tmp[e];
                if (key + 32 < Ntok) {       // prefetch next key block
                    __builtin_prefetch(kb + base + 32 * 768, 0, 0);
                    __builtin_prefetch(vb + base + 32 * 768, 0, 0);
                }
            } else {
                uint4 z = {0, 0, 0, 0};
                *(uint4*)&sK[r][c] = z;
                #pragma unroll
                for (int e = 0; e < 8; ++e) sVt[c + e][r] = f2bf(0.f);
            }
        }
        if (tid < 64) {
            int kk = tid >> 1, comp = tid & 1;
            int key = kb0 + kk;
            sPos[kk][comp] = (key < Ntok)
                ? positions[((size_t)(b * Ntok + key)) * 2 + comp] : 0.f;
        }
        asm volatile("s_wait_asynccnt 0x0" ::: "memory");
        __syncthreads();

        // S = Q @ K^T for 32 keys -> two 16x16 f32 fragments
        v8f sfrag[2];
        #pragma unroll
        for (int nf = 0; nf < 2; ++nf) {
            const __bf16* kr = &sK[nf * 16 + l16][0];
            v16bf b0 = frag16(kr + half * 8,      kr + 16 + half * 8);
            v16bf b1 = frag16(kr + 32 + half * 8, kr + 48 + half * 8);
            v8f z = {};
            z = __builtin_amdgcn_wmma_f32_16x16x32_bf16(false, qa0, false, b0, (short)0, z, false, false);
            z = __builtin_amdgcn_wmma_f32_16x16x32_bf16(false, qa1, false, b1, (short)0, z, false, false);
            sfrag[nf] = z;
        }

        // scale + 2-D ALiBi bias + masking
        #pragma unroll
        for (int nf = 0; nf < 2; ++nf) {
            int krow = kb0 + nf * 16 + l16;
            bool kv = krow < Ntok;
            int keepk = (kv && !(krow == 0 || (krow >= 1 && krow < 1 + nreg))) ? 1 : 0;
            float kpx = sPos[nf * 16 + l16][0];
            float kpy = sPos[nf * 16 + l16][1];
            #pragma unroll
            for (int r = 0; r < 8; ++r) {
                float sv;
                if (kv) {
                    float biasv = 0.f;
                    if (keepq[r] && keepk) {
                        float dx = qpx[r] - kpx, dy = qpy[r] - kpy;
                        biasv = -slope * sqrtf(dx * dx + dy * dy) / ps;
                    }
                    sv = sfrag[nf][r] * scale + biasv;
                } else {
                    sv = -1e30f;
                }
                sfrag[nf][r] = sv;
            }
        }

        // online softmax per row (reduce across the 16 lanes of each half)
        #pragma unroll
        for (int r = 0; r < 8; ++r) {
            float mx = fmaxf(sfrag[0][r], sfrag[1][r]);
            #pragma unroll
            for (int o = 8; o >= 1; o >>= 1) mx = fmaxf(mx, __shfl_xor(mx, o, 32));
            float mnew = fmaxf(m[r], mx);
            float corr = expf(m[r] - mnew);
            m[r] = mnew;
            float p0 = expf(sfrag[0][r] - mnew);
            float p1 = expf(sfrag[1][r] - mnew);
            float rs = p0 + p1;
            #pragma unroll
            for (int o = 8; o >= 1; o >>= 1) rs += __shfl_xor(rs, o, 32);
            l[r] = l[r] * corr + rs;
            #pragma unroll
            for (int t = 0; t < 4; ++t) accO[t][r] *= corr;
            sfrag[0][r] = p0;
            sfrag[1][r] = p1;
        }

        // P (C layout) -> LDS -> A fragment (bf16), then O += P @ V
        #pragma unroll
        for (int r = 0; r < 8; ++r) {
            sP[wave][r + 8 * half][l16]      = f2bf(sfrag[0][r]);
            sP[wave][r + 8 * half][16 + l16] = f2bf(sfrag[1][r]);
        }
        const __bf16* pr = &sP[wave][l16][0];
        v16bf pa = frag16(pr + half * 8, pr + 16 + half * 8);
        #pragma unroll
        for (int t = 0; t < 4; ++t) {
            const __bf16* vr = &sVt[t * 16 + l16][0];
            v16bf bv = frag16(vr + half * 8, vr + 16 + half * 8);
            accO[t] = __builtin_amdgcn_wmma_f32_16x16x32_bf16(
                false, pa, false, bv, (short)0, accO[t], false, false);
        }
    }

    // write O (bf16, token-major)
    #pragma unroll
    for (int r = 0; r < 8; ++r) {
        int row = qbase + r + 8 * half;
        if (row < Ntok) {
            float invl = 1.f / l[r];
            #pragma unroll
            for (int t = 0; t < 4; ++t) {
                ob[((size_t)(b * Ntok + row)) * 768 + h * HDIM + t * 16 + l16]
                    = f2bf(accO[t][r] * invl);
            }
        }
    }
}

// ---------------------------------------------------------------------------
// host-side orchestration
// ---------------------------------------------------------------------------
extern "C" void kernel_launch(void* const* d_in, const int* in_sizes, int n_in,
                              void* d_out, int out_size, void* d_ws, size_t ws_size,
                              hipStream_t stream) {
    const float* x     = (const float*)d_in[0];
    const float* pos   = (const float*)d_in[1];
    const float* psz   = (const float*)d_in[2];
    const int*   nreg  = (const int*)d_in[3];
    const float* ln1_g = (const float*)d_in[4];
    const float* ln1_b = (const float*)d_in[5];
    const float* wq    = (const float*)d_in[6];
    const float* bq    = (const float*)d_in[7];
    const float* wk    = (const float*)d_in[8];
    const float* bk    = (const float*)d_in[9];
    const float* wv    = (const float*)d_in[10];
    const float* bv    = (const float*)d_in[11];
    const float* wo    = (const float*)d_in[12];
    const float* bo    = (const float*)d_in[13];
    const float* qn_g  = (const float*)d_in[14];
    const float* qn_b  = (const float*)d_in[15];
    const float* kn_g  = (const float*)d_in[16];
    const float* kn_b  = (const float*)d_in[17];
    const float* ln2_g = (const float*)d_in[18];
    const float* ln2_b = (const float*)d_in[19];
    const float* w1    = (const float*)d_in[20];
    const float* b1    = (const float*)d_in[21];
    const float* w2    = (const float*)d_in[22];
    const float* b2    = (const float*)d_in[23];
    float* out = (float*)d_out;

    const int D  = 768;
    const int FF = 3072;
    const int Bsz  = in_sizes[2];             // 4
    const int MT   = in_sizes[0] / D;         // B*N = 4100
    const int Ntok = MT / Bsz;                // 1025

    char* ws = (char*)d_ws;
    // workspace layout (bytes, all 256-aligned)
    __bf16* wq_bf  = (__bf16*)(ws + 0);
    __bf16* wk_bf  = (__bf16*)(ws + 1179648);
    __bf16* wv_bf  = (__bf16*)(ws + 2359296);
    __bf16* wo_bf  = (__bf16*)(ws + 3538944);
    __bf16* w1_bf  = (__bf16*)(ws + 4718592);
    __bf16* w2_bf  = (__bf16*)(ws + 9437184);
    __bf16* xn_bf  = (__bf16*)(ws + 14155776);
    float*  q_raw  = (float*) (ws + 20453376);
    float*  x1     = (float*) (ws + 20453376);   // reuse q_raw after qk-norm
    float*  k_raw  = (float*) (ws + 33048576);
    __bf16* q_bf   = (__bf16*)(ws + 45643776);
    __bf16* h_bf   = (__bf16*)(ws + 45643776);   // reuse q/k/v/o bf16 for MLP hidden
    __bf16* k_bf   = (__bf16*)(ws + 51941376);
    __bf16* v_bf   = (__bf16*)(ws + 58238976);
    __bf16* o_bf   = (__bf16*)(ws + 64536576);
    __bf16* xn2_bf = (__bf16*)(ws + 70834176);

    // 1. weights fp32 -> bf16
    k_cvt<<<(D * D + 255) / 256, 256, 0, stream>>>(wq, wq_bf, D * D);
    k_cvt<<<(D * D + 255) / 256, 256, 0, stream>>>(wk, wk_bf, D * D);
    k_cvt<<<(D * D + 255) / 256, 256, 0, stream>>>(wv, wv_bf, D * D);
    k_cvt<<<(D * D + 255) / 256, 256, 0, stream>>>(wo, wo_bf, D * D);
    k_cvt<<<(D * FF + 255) / 256, 256, 0, stream>>>(w1, w1_bf, D * FF);
    k_cvt<<<(FF * D + 255) / 256, 256, 0, stream>>>(w2, w2_bf, FF * D);

    // 2. LN1
    k_ln<<<MT, 256, 0, stream>>>(x, ln1_g, ln1_b, xn_bf, D);

    dim3 g768((MT + BM - 1) / BM, D / BN);
    dim3 gFF((MT + BM - 1) / BM, FF / BN);

    // 3. QKV projections
    k_gemm<<<g768, 256, 0, stream>>>(xn_bf, wq_bf, bq, nullptr, q_raw, nullptr, MT, D, D, 0);
    k_gemm<<<g768, 256, 0, stream>>>(xn_bf, wk_bf, bk, nullptr, k_raw, nullptr, MT, D, D, 0);
    k_gemm<<<g768, 256, 0, stream>>>(xn_bf, wv_bf, bv, nullptr, nullptr, v_bf, MT, D, D, 0);

    // 4. qk-norm over head_dim (rows of 64)
    k_ln<<<MT * HEADS, 64, 0, stream>>>(q_raw, qn_g, qn_b, q_bf, HDIM);
    k_ln<<<MT * HEADS, 64, 0, stream>>>(k_raw, kn_g, kn_b, k_bf, HDIM);

    // 5. flash attention with 2-D ALiBi
    dim3 ga(Bsz * HEADS, (Ntok + 127) / 128);
    k_attn<<<ga, 256, 0, stream>>>(q_bf, k_bf, v_bf, pos, psz, nreg, o_bf, Ntok);

    // 6. output projection + residual -> x1
    k_gemm<<<g768, 256, 0, stream>>>(o_bf, wo_bf, bo, x, x1, nullptr, MT, D, D, 0);

    // 7. LN2
    k_ln<<<MT, 256, 0, stream>>>(x1, ln2_g, ln2_b, xn2_bf, D);

    // 8. MLP: gelu(xn2 @ w1 + b1) -> h ; h @ w2 + b2 + x1 -> out
    k_gemm<<<gFF, 256, 0, stream>>>(xn2_bf, w1_bf, b1, nullptr, nullptr, h_bf, MT, FF, D, 1);
    k_gemm<<<g768, 256, 0, stream>>>(h_bf, w2_bf, b2, x1, out, nullptr, MT, D, FF, 0);
}
